// ExactSizePatchNetworkONNX_16028817949425
// MI455X (gfx1250) — matrix operand order repaired and checked by reading, not compiled, not run on hardware
//
#include <hip/hip_runtime.h>
#include <hip/hip_bf16.h>

typedef __attribute__((ext_vector_type(16))) _Float16 v16h;
typedef __attribute__((ext_vector_type(8)))  float    v8f;
typedef __attribute__((ext_vector_type(4)))  unsigned int v4u;
typedef __attribute__((ext_vector_type(4)))  int      v4i;
typedef __attribute__((ext_vector_type(8)))  int      v8i;

#define HH 1024
#define WW 1024
#define CC 16
#define BB 2
#define PATCH 256
#define NSTART 5      // patch starts at 0,192,384,576,768
#define STRIDE 192
#define KTOT 144      // 16 ci * 9 taps
#define SCOLS 264     // staged cols (258 used, padded)

// dynamic LDS layout (bytes)
#define RAW_OFF   0
#define RAW_BYTES (CC * 6 * SCOLS * 4)            // 101376
#define F16_OFF   RAW_BYTES
#define F16_BYTES (CC * 6 * SCOLS * 2)            // 50688
#define GR_OFF    (F16_OFF + F16_BYTES)
#define GC_OFF    (GR_OFF + 32)
#define SMEM_BYTES (GC_OFF + SCOLS * 4)           // ~153 KB of the 320 KB WGP LDS

#if defined(__has_builtin)
#  if __has_builtin(__builtin_amdgcn_tensor_load_to_lds) && \
      __has_builtin(__builtin_amdgcn_s_wait_tensorcnt)
#    define HAVE_TDM 1
#  endif
#endif
#ifndef HAVE_TDM
#  define HAVE_TDM 0
#endif

union AFrag { v16h v; _Float16 h[16]; };
union CAcc  { v8f  v; float    f[8];  };

// LDS half-offset for reduction index K = ci*9 + dy*3 + dx ; -1 => zero pad
__host__ __device__ constexpr int ldsOffHalf(int K) {
    return (K >= KTOT) ? -1
         : ((K / 9) * (6 * SCOLS) + ((K % 9) / 3) * SCOLS + ((K % 9) % 3));
}

__device__ __forceinline__ int coverCnt(int v) {
    int c = 0;
    #pragma unroll
    for (int i = 0; i < NSTART; ++i)
        c += (v >= i * STRIDE && v <= i * STRIDE + PATCH - 1) ? 1 : 0;
    return c;
}

// ---------------- zero d_out (vectorized) ----------------
__global__ void zero_kernel(float4* out, int n4) {
    int i = blockIdx.x * blockDim.x + threadIdx.x;
    if (i < n4) out[i] = make_float4(0.f, 0.f, 0.f, 0.f);
}

// ---------------- patch conv: TDM stage -> masked f16 -> WMMA -> overlap-add --
// grid: (64 row-chunks of 4 rows, 25 patches, 2 batches), block: 256 threads
__global__ __launch_bounds__(256)
void patch_conv_wmma(const float* __restrict__ x,
                     const float* __restrict__ wgt,   // [co][K], K = ci*9+dy*3+dx
                     float* __restrict__ out) {
    extern __shared__ char smem[];
    float*    sRaw = (float*)(smem + RAW_OFF);        // DMA landing pad (f32)
    _Float16* sF16 = (_Float16*)(smem + F16_OFF);     // masked f16 tile
    float*    sGr  = (float*)(smem + GR_OFF);
    float*    sGc  = (float*)(smem + GC_OFF);

    const int tid    = threadIdx.x;
    const int lane   = tid & 31;
    const int hi     = lane >> 4;
    const int wave   = tid >> 5;
    const int m      = lane & 15;
    const int r0     = blockIdx.x * 4;
    const int pid    = blockIdx.y;
    const int bidx   = blockIdx.z;
    const int hstart = (pid / NSTART) * STRIDE;
    const int wstart = (pid % NSTART) * STRIDE;

    // clamped tile geometry (uniform): image rows/cols actually fetched
    const int gr0   = hstart + r0 - 1;
    const int rowlo = gr0 < 0 ? 0 : gr0;
    const int lrOff = rowlo - gr0;                         // 0 or 1
    const int nrows = min(6 - lrOff, HH - rowlo);
    const int gc0   = wstart - 1;
    const int collo = gc0 < 0 ? 0 : gc0;
    const int lcOff = collo - gc0;                         // 0 or 1
    const int ncols = min(SCOLS - lcOff, WW - collo);

#if HAVE_TDM
    // ---- issue TDM 3-D tile load (ci x rows x cols, f32) from wave 0 ----
    if (wave == 0) {
        unsigned long long gaddr = (unsigned long long)(uintptr_t)x
            + 4ull * (((unsigned long long)(bidx * CC)) * (HH * WW)
                      + (unsigned long long)rowlo * WW + collo);
        unsigned ldsA = (unsigned)(uintptr_t)(void*)sRaw;  // low 32b = LDS offset
        v4u g0 = { 1u,                                   // count=1, user mode
                   ldsA,
                   (unsigned)(gaddr & 0xFFFFFFFFu),
                   (unsigned)((gaddr >> 32) & 0x1FFFFFFu) | (2u << 30) }; // type=2
        v8i g1 = { (int)(2u << 16),                      // wg_mask=0, data_size=4B
                   (int)((unsigned)(ncols & 0xFFFF) << 16),      // tensor_dim0 lo16
                   (int)((unsigned)(nrows & 0xFFFF) << 16),      // dim0 hi=0, dim1 lo16
                   (int)((unsigned)(ncols & 0xFFFF) << 16),      // dim1 hi=0, tile_dim0
                   (int)((unsigned)(nrows & 0xFFFF)
                         | ((unsigned)CC << 16)),                // tile_dim1, tile_dim2
                   (int)WW,                                      // dim0 stride = 1024
                   0,                                            // stride0 hi, stride1 lo16=0
                   (int)((HH * WW) >> 16) };                     // stride1[47:16] = 16
        v4i g2 = { CC, 0, (int)(HH * WW), 0 };           // tensor_dim2, dim2 stride
        v4i g3 = { 0, 0, 0, 0 };
#if defined(__clang_major__) && (__clang_major__ >= 23)
        v8i g4 = { 0, 0, 0, 0, 0, 0, 0, 0 };
        __builtin_amdgcn_tensor_load_to_lds(g0, g1, g2, g3, g4, 0);
#else
        __builtin_amdgcn_tensor_load_to_lds(g0, g1, g2, g3, 0);
#endif
    }
#endif

    // ---- gauss tables built while the DMA is in flight ----
    const float sc = 2.0f / 255.0f;
    if (tid < 6) {
        int pr = r0 - 1 + tid;
        float la = -1.0f + sc * (float)pr;
        sGr[tid] = (pr >= 0 && pr < PATCH) ? __expf(-(la * la) * (1.0f / 1.5f)) : 0.0f;
    }
    for (int j = tid; j < SCOLS; j += 256) {
        int pc = j - 1;
        float lb = -1.0f + sc * (float)pc;
        sGc[j] = (pc >= 0 && pc < PATCH) ? __expf(-(lb * lb) * (1.0f / 1.5f)) : 0.0f;
    }

    // ---- weight A-fragments (16 x 160, zero-padded), ISA 16-bit A layout ----
    AFrag afrag[5];
    #pragma unroll
    for (int kk = 0; kk < 5; ++kk) {
        #pragma unroll
        for (int e = 0; e < 16; ++e) {
            int K = 32 * kk + (hi == 0 ? (e < 8 ? e : e + 8)
                                       : (e < 8 ? e + 8 : e + 16));
            float wv = (K < KTOT) ? wgt[m * KTOT + K] : 0.0f;
            afrag[kk].h[e] = (_Float16)wv;
        }
    }

#if HAVE_TDM
    if (wave == 0) __builtin_amdgcn_s_wait_tensorcnt(0);   // issuing wave's counter
    __syncthreads();                                        // data + tables visible
    // ---- masked f32 -> f16 conversion (LDS -> LDS) ----
    const int planeSz = nrows * ncols;
    for (int i = tid; i < CC * 6 * SCOLS; i += 256) {
        int ci  = i / (6 * SCOLS);
        int rem = i % (6 * SCOLS);
        int lr  = rem / SCOLS;
        int lc  = rem % SCOLS;
        bool inR = (lr >= lrOff) && (lr < lrOff + nrows)
                && (lc >= lcOff) && (lc < lcOff + ncols);
        int idx = ci * planeSz + (lr - lrOff) * ncols + (lc - lcOff);
        float raw = sRaw[inR ? idx : 0];
        float v = (inR ? raw : 0.0f) * sGr[lr] * sGc[lc];  // mask zeroes the halo
        sF16[i] = (_Float16)v;
    }
#else
    __syncthreads();
    // ---- fallback: direct masked staging (clamped coalesced loads) ----
    for (int i = tid; i < CC * 6 * SCOLS; i += 256) {
        int ci  = i / (6 * SCOLS);
        int rem = i % (6 * SCOLS);
        int lr  = rem / SCOLS;
        int lc  = rem % SCOLS;
        int prc = min(max(r0 - 1 + lr, 0), PATCH - 1);
        int pcc = min(max(lc - 1, 0), PATCH - 1);
        size_t gidx = ((size_t)(bidx * CC + ci)) * (HH * WW)
                    + (size_t)(hstart + prc) * WW + (wstart + pcc);
        float v = x[gidx] * sGr[lr] * sGc[lc];
        sF16[i] = (_Float16)v;
    }
#endif
    __syncthreads();

    // ---- 64 output tiles (4 rows x 16 col-tiles), 8 per wave ----
    for (int t = wave; t < 64; t += 8) {
        const int rr = t >> 4;
        const int ct = t & 15;
        const _Float16* sBase = sF16 + (rr * SCOLS + ct * 16 + m);
        CAcc c; c.v = (v8f){};

        #pragma unroll
        for (int kk = 0; kk < 5; ++kk) {
            AFrag b;  // B (32x16): lanes 0-15 K=e<8?e:e+8, lanes 16-31 K=e<8?e+8:e+16; N=lane&15
            #pragma unroll
            for (int e = 0; e < 16; ++e) {
                int off  = hi ? ldsOffHalf(32 * kk + (e < 8 ? e + 8 : e + 16))
                              : ldsOffHalf(32 * kk + (e < 8 ? e : e + 8));
                int offc = off < 0 ? 0 : off;
                _Float16 v = sBase[offc];
                b.h[e] = (off < 0) ? (_Float16)0.0f : v;
            }
            c.v = __builtin_amdgcn_wmma_f32_16x16x32_f16(
                false, afrag[kk].v, false, b.v, (short)0, c.v, false, false);
        }

        // ---- overlap-add: C/D layout VGPR r -> M = r + 8*hi, N = lane&15 ----
        const int gh  = hstart + r0 + rr;
        const int gw0 = wstart + ct * 16;
        const int gw  = gw0 + m;
        const bool excl = (coverCnt(gh) == 1) && (coverCnt(gw0) == 1)
                       && (coverCnt(gw0 + 15) == 1);   // wave-uniform exclusive tile
        #pragma unroll
        for (int r = 0; r < 8; ++r) {
            int co = r + 8 * hi;
            float* p = out + ((size_t)(bidx * CC + co)) * (HH * WW)
                           + (size_t)gh * WW + gw;
            if (excl) {
                *p = c.f[r];
            } else {
                __hip_atomic_fetch_add(p, c.f[r], __ATOMIC_RELAXED,
                                       __HIP_MEMORY_SCOPE_AGENT);
            }
        }
    }
}

// ---------------- normalize by analytic coverage, add bias ----------------
__global__ void norm_kernel(float* out, const float* __restrict__ bias) {
    int idx = blockIdx.x * blockDim.x + threadIdx.x;
    if (idx >= HH * WW) return;
    int h = idx >> 10;
    int w = idx & 1023;
    float ws  = (float)(coverCnt(h) * coverCnt(w));
    float inv = 1.0f / (ws + 1e-8f);
    #pragma unroll
    for (int b = 0; b < BB; ++b) {
        #pragma unroll
        for (int c = 0; c < CC; ++c) {
            size_t o = ((size_t)(b * CC + c) << 20) + (size_t)idx;
            out[o] = (out[o] + ws * bias[c]) * inv;
        }
    }
}

extern "C" void kernel_launch(void* const* d_in, const int* in_sizes, int n_in,
                              void* d_out, int out_size, void* d_ws, size_t ws_size,
                              hipStream_t stream) {
    (void)in_sizes; (void)n_in; (void)d_ws; (void)ws_size;
    const float* x      = (const float*)d_in[0];
    const float* conv_w = (const float*)d_in[1];
    const float* conv_b = (const float*)d_in[2];
    float*       out    = (float*)d_out;

    {   // 1) zero accumulator (harness poisons d_out; must re-zero every call)
        int n4 = out_size / 4;
        zero_kernel<<<(n4 + 255) / 256, 256, 0, stream>>>((float4*)out, n4);
    }
    {   // 2) per-patch masked conv + overlap-add (TDM-staged, WMMA)
        dim3 grid(PATCH / 4, NSTART * NSTART, BB);   // 64 x 25 x 2
        patch_conv_wmma<<<grid, 256, SMEM_BYTES, stream>>>(x, conv_w, out);
    }
    {   // 3) normalize
        int n = HH * WW;
        norm_kernel<<<(n + 255) / 256, 256, 0, stream>>>(out, conv_b);
    }
}
